// SpectralWormhole_56762287784329
// MI455X (gfx1250) — matrix operand order repaired and checked by reading, not compiled, not run on hardware
//
#include <hip/hip_runtime.h>
#include <hip/hip_bf16.h>

typedef __attribute__((ext_vector_type(16))) _Float16 v16h;
typedef __attribute__((ext_vector_type(8)))  _Float16 v8h;
typedef __attribute__((ext_vector_type(8)))  float    v8f;

#define NBANDS 7
#define BQ 4
#define TSEQ 2048
#define DDIM 256
#define MROWS (BQ * TSEQ)   // 8192 rows per band
#define TOPK 16

__device__ __forceinline__ float waveReduceAdd(float x) {
#pragma unroll
  for (int m = 16; m >= 1; m >>= 1) x += __shfl_xor(x, m, 32);
  return x;
}

// ---------------------------------------------------------------------------
// LayerNorm: one wave per row of [7*8192, 256]; writes f16 normed activations.
// ---------------------------------------------------------------------------
__global__ void __launch_bounds__(256)
ln_to_f16_kernel(const float* __restrict__ bands,
                 const float* __restrict__ g,
                 const float* __restrict__ bta,
                 _Float16* __restrict__ Y) {
  int lane = threadIdx.x & 31;
  int wv = threadIdx.x >> 5;
  size_t row = (size_t)blockIdx.x * 8 + wv;       // 0 .. 57343
  int band = (int)(row >> 13);                    // row / 8192
  const float4* p = (const float4*)(bands + row * DDIM + lane * 8);
  float4 x0 = p[0], x1 = p[1];
  float s = x0.x + x0.y + x0.z + x0.w + x1.x + x1.y + x1.z + x1.w;
  float ss = x0.x * x0.x + x0.y * x0.y + x0.z * x0.z + x0.w * x0.w +
             x1.x * x1.x + x1.y * x1.y + x1.z * x1.z + x1.w * x1.w;
  s = waveReduceAdd(s);
  ss = waveReduceAdd(ss);
  float mean = s * (1.0f / 256.0f);
  float var = ss * (1.0f / 256.0f) - mean * mean;
  float rstd = rsqrtf(var + 1e-5f);
  const float4* gp = (const float4*)(g + band * DDIM + lane * 8);
  const float4* bp = (const float4*)(bta + band * DDIM + lane * 8);
  float4 g0 = gp[0], g1 = gp[1], b0 = bp[0], b1 = bp[1];
  v8h out;
  out[0] = (_Float16)((x0.x - mean) * rstd * g0.x + b0.x);
  out[1] = (_Float16)((x0.y - mean) * rstd * g0.y + b0.y);
  out[2] = (_Float16)((x0.z - mean) * rstd * g0.z + b0.z);
  out[3] = (_Float16)((x0.w - mean) * rstd * g0.w + b0.w);
  out[4] = (_Float16)((x1.x - mean) * rstd * g1.x + b1.x);
  out[5] = (_Float16)((x1.y - mean) * rstd * g1.y + b1.y);
  out[6] = (_Float16)((x1.z - mean) * rstd * g1.z + b1.z);
  out[7] = (_Float16)((x1.w - mean) * rstd * g1.w + b1.w);
  *(v8h*)(Y + row * DDIM + lane * 8) = out;
}

// ---------------------------------------------------------------------------
// Row convert f32 -> f16, optional L2 normalization or constant scale.
// One wave per row of [8192,256].
// ---------------------------------------------------------------------------
__global__ void __launch_bounds__(256)
rows_to_f16_kernel(const float* __restrict__ X, _Float16* __restrict__ Y,
                   int doNorm, float scale) {
  int lane = threadIdx.x & 31;
  int wv = threadIdx.x >> 5;
  size_t row = (size_t)blockIdx.x * 8 + wv;       // 0 .. 8191
  const float4* p = (const float4*)(X + row * DDIM + lane * 8);
  float4 x0 = p[0], x1 = p[1];
  float sc = scale;
  if (doNorm) {
    float ss = x0.x * x0.x + x0.y * x0.y + x0.z * x0.z + x0.w * x0.w +
               x1.x * x1.x + x1.y * x1.y + x1.z * x1.z + x1.w * x1.w;
    ss = waveReduceAdd(ss);
    float n = sqrtf(ss);
    sc = 1.0f / fmaxf(n, 1e-12f);
  }
  v8h out;
  out[0] = (_Float16)(x0.x * sc);
  out[1] = (_Float16)(x0.y * sc);
  out[2] = (_Float16)(x0.z * sc);
  out[3] = (_Float16)(x0.w * sc);
  out[4] = (_Float16)(x1.x * sc);
  out[5] = (_Float16)(x1.y * sc);
  out[6] = (_Float16)(x1.z * sc);
  out[7] = (_Float16)(x1.w * sc);
  *(v8h*)(Y + row * DDIM + lane * 8) = out;
}

// ---------------------------------------------------------------------------
// Pack weight W[256,256] f32 into WMMA B-fragment layout, f16.
// Fragment (nt,kc), lane L, element j holds W[kc*32 + (L>=16?8:0) + (j<8 ? j : 8+j)][nt*16 + (L&15)]
// 128 frags * 32 lanes * 16 halves, stored contiguously for 32B vector loads.
// ---------------------------------------------------------------------------
__global__ void __launch_bounds__(256)
prep_weight_kernel(const float* __restrict__ W, _Float16* __restrict__ Wf) {
  int tid = blockIdx.x * 256 + threadIdx.x;   // 0..4095
  int frag = tid >> 5;                        // 0..127
  int lane = tid & 31;
  int nt = frag >> 3, kc = frag & 7;
  int k0 = kc * 32 + ((lane & 16) ? 8 : 0);
  int col = nt * 16 + (lane & 15);
  v16h t;
#pragma unroll
  for (int j = 0; j < 16; ++j) t[j] = (_Float16)W[(size_t)(k0 + (j < 8 ? j : j + 8)) * DDIM + col];
  *(v16h*)(Wf + ((size_t)frag * 32 + lane) * 16) = t;
}

// ---------------------------------------------------------------------------
// C[8192,256] = A_h[8192,256](f16) @ W(packed frags) + bias; compile-time +=.
// One wave per 32-row slab (2 WMMA row tiles): 16 resident A fragments,
// each B fragment load feeds 2 WMMAs (halves L2 weight traffic per FLOP).
// ---------------------------------------------------------------------------
template <int ACCUM>
__global__ void __launch_bounds__(256)
gemm256_wmma_kernel(const _Float16* __restrict__ A,
                    const _Float16* __restrict__ Wf,
                    const float* __restrict__ bias,
                    float* __restrict__ C) {
  int lane = threadIdx.x & 31;
  int wv = threadIdx.x >> 5;
  int slab = blockIdx.x * 8 + wv;             // 0..255
  int m0 = slab * 32;
  int mrow0 = m0 + (lane & 15);
  int mrow1 = m0 + 16 + (lane & 15);
  int abase = (lane & 16) ? 8 : 0;
  v16h a0[8], a1[8];
#pragma unroll
  for (int kc = 0; kc < 8; ++kc) {
    v8h lo0 = *(const v8h*)(A + (size_t)mrow0 * DDIM + kc * 32 + abase);
    v8h hi0 = *(const v8h*)(A + (size_t)mrow0 * DDIM + kc * 32 + abase + 16);
    v8h lo1 = *(const v8h*)(A + (size_t)mrow1 * DDIM + kc * 32 + abase);
    v8h hi1 = *(const v8h*)(A + (size_t)mrow1 * DDIM + kc * 32 + abase + 16);
#pragma unroll
    for (int e = 0; e < 8; ++e) {
      a0[kc][e] = lo0[e]; a0[kc][e + 8] = hi0[e];
      a1[kc][e] = lo1[e]; a1[kc][e + 8] = hi1[e];
    }
  }
  int rhi = (lane & 16) ? 8 : 0;
  for (int nt = 0; nt < 16; ++nt) {
    v8f acc0 = {}, acc1 = {};
#pragma unroll
    for (int kc = 0; kc < 8; ++kc) {
      v16h b = *(const v16h*)(Wf + ((size_t)(nt * 8 + kc) * 32 + lane) * 16);
      acc0 = __builtin_amdgcn_wmma_f32_16x16x32_f16(false, a0[kc], false, b,
                                                    (short)0, acc0, false, false);
      acc1 = __builtin_amdgcn_wmma_f32_16x16x32_f16(false, a1[kc], false, b,
                                                    (short)0, acc1, false, false);
    }
    int col = nt * 16 + (lane & 15);
    float bb = bias[col];
    float* c0 = C + (size_t)(m0 + rhi) * DDIM + col;
    float* c1 = C + (size_t)(m0 + 16 + rhi) * DDIM + col;
#pragma unroll
    for (int r = 0; r < 8; ++r) {
      float v0 = acc0[r] + bb;
      float v1 = acc1[r] + bb;
      if constexpr (ACCUM) {
        c0[(size_t)r * DDIM] += v0;
        c1[(size_t)r * DDIM] += v1;
      } else {
        c0[(size_t)r * DDIM] = v0;
        c1[(size_t)r * DDIM] = v1;
      }
    }
  }
}

// ---------------------------------------------------------------------------
// Fused cosine-sim / top-16 / softmax / V-gather attention.
// One wave per 16-query tile within one batch; streams 128 key tiles via WMMA.
// Key tile's 8 B fragments are preloaded into registers as one 16-load clause
// so the WMMA chain overlaps the tail of the load latency (partial loadcnt
// waits instead of a full drain per WMMA).
// ---------------------------------------------------------------------------
__global__ void __launch_bounds__(256)
topk_attn_wmma_kernel(const _Float16* __restrict__ Qn,
                      const _Float16* __restrict__ Kn,
                      const float* __restrict__ V,
                      const float* __restrict__ temps, int tIdx,
                      float* __restrict__ O) {
  __shared__ float sw[8][16][16];
  __shared__ int   si[8][16][16];
  int lane = threadIdx.x & 31;
  int wv = threadIdx.x >> 5;
  int gw = blockIdx.x * 8 + wv;               // 0..511
  int bb = gw >> 7;                           // batch 0..3
  int qt = gw & 127;
  int m0 = bb * TSEQ + qt * 16;
  float invT = 1.0f / temps[tIdx];

  // A fragments: 16 query rows, K-dim 256 in 8 chunks of 32
  int abase = (lane & 16) ? 8 : 0;
  int mrow = m0 + (lane & 15);
  v16h a[8];
#pragma unroll
  for (int kc = 0; kc < 8; ++kc) {
    v8h lo = *(const v8h*)(Qn + (size_t)mrow * DDIM + kc * 32 + abase);
    v8h hi = *(const v8h*)(Qn + (size_t)mrow * DDIM + kc * 32 + abase + 16);
#pragma unroll
    for (int e = 0; e < 8; ++e) { a[kc][e] = lo[e]; a[kc][e + 8] = hi[e]; }
  }

  float vals[TOPK];
  int keys[TOPK];
#pragma unroll
  for (int t = 0; t < TOPK; ++t) { vals[t] = -3.0e38f; keys[t] = 0; }
  float vmin = -3.0e38f;
  int imin = 0;

  int bcol = (lane & 16) ? 8 : 0;
  const _Float16* kb = Kn + (size_t)bb * TSEQ * DDIM;
  int rhi = (lane & 16) ? 8 : 0;

  for (int st = 0; st < 128; ++st) {
    if (st < 127)
      __builtin_prefetch(kb + (size_t)((st + 1) * 16 + (lane & 15)) * DDIM, 0, 1);
    const _Float16* krow = kb + (size_t)(st * 16 + (lane & 15)) * DDIM + bcol;

    // Preload the whole key tile (8 fragments) as one load clause.
    v16h bfr[8];
#pragma unroll
    for (int kc = 0; kc < 8; ++kc) {
      v8h blo = *(const v8h*)(krow + kc * 32);
      v8h bhi = *(const v8h*)(krow + kc * 32 + 16);
      for (int e = 0; e < 8; ++e) { bfr[kc][e] = blo[e]; bfr[kc][e + 8] = bhi[e]; }
    }

    v8f acc = {};
#pragma unroll
    for (int kc = 0; kc < 8; ++kc)
      acc = __builtin_amdgcn_wmma_f32_16x16x32_f16(false, a[kc], false, bfr[kc],
                                                   (short)0, acc, false, false);

#pragma unroll
    for (int r = 0; r < 8; ++r)
      sw[wv][r + rhi][lane & 15] = acc[r] * invT;
    __syncthreads();
    if (lane < 16) {
#pragma unroll
      for (int j = 0; j < 16; ++j) {
        float c = sw[wv][lane][j];
        if (c > vmin) {
          int key = st * 16 + j;
#pragma unroll
          for (int t = 0; t < TOPK; ++t)
            if (t == imin) { vals[t] = c; keys[t] = key; }
          vmin = vals[0]; imin = 0;
#pragma unroll
          for (int t = 1; t < TOPK; ++t)
            if (vals[t] < vmin) { vmin = vals[t]; imin = t; }
        }
      }
    }
    __syncthreads();
  }

  // softmax over top-16, stage weights + indices to LDS
  if (lane < 16) {
    float mx = vals[0];
#pragma unroll
    for (int t = 1; t < TOPK; ++t) mx = fmaxf(mx, vals[t]);
    float w[TOPK];
    float s = 0.0f;
#pragma unroll
    for (int t = 0; t < TOPK; ++t) { w[t] = __expf(vals[t] - mx); s += w[t]; }
    float inv = 1.0f / s;
#pragma unroll
    for (int t = 0; t < TOPK; ++t) {
      sw[wv][lane][t] = w[t] * inv;
      si[wv][lane][t] = keys[t];
    }
  }
  __syncthreads();

  // wave-cooperative gather: per query row, 16 weighted V rows, 8 dims/lane
  const float* vb = V + (size_t)bb * TSEQ * DDIM;
  for (int qr = 0; qr < 16; ++qr) {
    float acc[8];
#pragma unroll
    for (int e = 0; e < 8; ++e) acc[e] = 0.0f;
#pragma unroll 4
    for (int j = 0; j < 16; ++j) {
      float w = sw[wv][qr][j];
      int key = si[wv][qr][j];
      const float4* vp = (const float4*)(vb + (size_t)key * DDIM + lane * 8);
      float4 x0 = vp[0], x1 = vp[1];
      acc[0] += w * x0.x; acc[1] += w * x0.y; acc[2] += w * x0.z; acc[3] += w * x0.w;
      acc[4] += w * x1.x; acc[5] += w * x1.y; acc[6] += w * x1.z; acc[7] += w * x1.w;
    }
    float4 o0, o1;
    o0.x = acc[0]; o0.y = acc[1]; o0.z = acc[2]; o0.w = acc[3];
    o1.x = acc[4]; o1.y = acc[5]; o1.z = acc[6]; o1.w = acc[7];
    float* op = O + (size_t)(m0 + qr) * DDIM + lane * 8;
    *(float4*)op = o0;
    *(float4*)(op + 4) = o1;
  }
}

// ---------------------------------------------------------------------------
// Elementwise helpers (grid-stride, float4)
// ---------------------------------------------------------------------------
__global__ void ew_copy4_kernel(const float4* __restrict__ s, float4* __restrict__ d, int n4) {
  for (int i = blockIdx.x * blockDim.x + threadIdx.x; i < n4; i += gridDim.x * blockDim.x)
    d[i] = s[i];
}
__global__ void ew_zero4_kernel(float4* __restrict__ d, int n4) {
  for (int i = blockIdx.x * blockDim.x + threadIdx.x; i < n4; i += gridDim.x * blockDim.x) {
    float4 z; z.x = 0.f; z.y = 0.f; z.z = 0.f; z.w = 0.f;
    d[i] = z;
  }
}
__global__ void ew_add4_kernel(float4* __restrict__ d, const float4* __restrict__ s, int n4) {
  for (int i = blockIdx.x * blockDim.x + threadIdx.x; i < n4; i += gridDim.x * blockDim.x) {
    float4 a = d[i], b = s[i];
    a.x += b.x; a.y += b.y; a.z += b.z; a.w += b.w;
    d[i] = a;
  }
}

// ---------------------------------------------------------------------------
// Host orchestration
// ---------------------------------------------------------------------------
extern "C" void kernel_launch(void* const* d_in, const int* in_sizes, int n_in,
                              void* d_out, int out_size, void* d_ws, size_t ws_size,
                              hipStream_t stream) {
  const float* bands     = (const float*)d_in[0];
  const float* pair_qW   = (const float*)d_in[1];
  const float* pair_qb   = (const float*)d_in[2];
  const float* pair_kW   = (const float*)d_in[3];
  const float* pair_kb   = (const float*)d_in[4];
  const float* pair_vW   = (const float*)d_in[5];
  const float* pair_vb   = (const float*)d_in[6];
  const float* pair_oW   = (const float*)d_in[7];
  const float* pair_ob   = (const float*)d_in[8];
  const float* bridge_qW = (const float*)d_in[9];
  const float* bridge_qb = (const float*)d_in[10];
  const float* bridge_kW = (const float*)d_in[11];
  const float* bridge_kb = (const float*)d_in[12];
  const float* bridge_vW = (const float*)d_in[13];
  const float* bridge_vb = (const float*)d_in[14];
  const float* bridge_oW = (const float*)d_in[15];
  const float* bridge_ob = (const float*)d_in[16];
  const float* temps     = (const float*)d_in[17];
  const float* ln_g      = (const float*)d_in[18];
  const float* ln_b      = (const float*)d_in[19];
  float* outp = (float*)d_out;

  const size_t BAND = (size_t)MROWS * DDIM;   // 2,097,152 elements per band

  char* ws = (char*)d_ws;
  size_t off = 0;
  auto walloc = [&](size_t bytes) -> void* {
    void* p = ws + off;
    off = (off + bytes + 255) & ~(size_t)255;
    return p;
  };
  _Float16* normed_h = (_Float16*)walloc((size_t)NBANDS * BAND * 2);
  float*    qf       = (float*)walloc(BAND * 4);
  float*    kf       = (float*)walloc(BAND * 4);
  float*    vf       = (float*)walloc(BAND * 4);
  _Float16* qn       = (_Float16*)walloc(BAND * 2);
  _Float16* kn       = (_Float16*)walloc(BAND * 2);
  float*    att      = (float*)walloc(BAND * 4);
  float*    attacc   = (float*)walloc(BAND * 4);
  _Float16* atth     = (_Float16*)walloc(BAND * 2);
  _Float16* wh       = (_Float16*)walloc((size_t)DDIM * DDIM * 2);

  // out = bands (residual init)
  {
    int n4 = (int)((size_t)NBANDS * BAND / 4);
    ew_copy4_kernel<<<4096, 256, 0, stream>>>((const float4*)bands, (float4*)outp, n4);
  }
  // LayerNorm all bands -> f16
  ln_to_f16_kernel<<<(NBANDS * MROWS) / 8, 256, 0, stream>>>(bands, ln_g, ln_b, normed_h);

  auto gemm = [&](const _Float16* A, const float* W, const float* bias,
                  float* C, int accum) {
    prep_weight_kernel<<<16, 256, 0, stream>>>(W, wh);
    if (accum)
      gemm256_wmma_kernel<1><<<32, 256, 0, stream>>>(A, wh, bias, C);
    else
      gemm256_wmma_kernel<0><<<32, 256, 0, stream>>>(A, wh, bias, C);
  };

  const int SRC[6] = {0, 6, 1, 5, 2, 4};
  const int DST[6] = {6, 0, 5, 1, 4, 2};
  const size_t WSTRIDE = (size_t)DDIM * DDIM;

  for (int pidx = 0; pidx < 6; ++pidx) {
    const _Float16* Asrc = normed_h + (size_t)SRC[pidx] * BAND;
    const _Float16* Adst = normed_h + (size_t)DST[pidx] * BAND;
    gemm(Asrc, pair_qW + pidx * WSTRIDE, pair_qb + pidx * DDIM, qf, 0);
    gemm(Adst, pair_kW + pidx * WSTRIDE, pair_kb + pidx * DDIM, kf, 0);
    gemm(Adst, pair_vW + pidx * WSTRIDE, pair_vb + pidx * DDIM, vf, 0);
    rows_to_f16_kernel<<<MROWS / 8, 256, 0, stream>>>(qf, qn, 1, 1.0f);
    rows_to_f16_kernel<<<MROWS / 8, 256, 0, stream>>>(kf, kn, 1, 1.0f);
    topk_attn_wmma_kernel<<<64, 256, 0, stream>>>(qn, kn, vf, temps, pidx, att);
    rows_to_f16_kernel<<<MROWS / 8, 256, 0, stream>>>(att, atth, 0, 1.0f);
    gemm(atth, pair_oW + pidx * WSTRIDE, pair_ob + pidx * DDIM,
         outp + (size_t)SRC[pidx] * BAND, 1);
  }

  // Bridge: q from band 3; attend over the 6 other bands, averaged.
  gemm(normed_h + (size_t)3 * BAND, bridge_qW, bridge_qb, qf, 0);
  rows_to_f16_kernel<<<MROWS / 8, 256, 0, stream>>>(qf, qn, 1, 1.0f);
  {
    int n4 = (int)(BAND / 4);
    ew_zero4_kernel<<<2048, 256, 0, stream>>>((float4*)attacc, n4);
  }
  const int OTH[6] = {0, 1, 2, 4, 5, 6};
  for (int i = 0; i < 6; ++i) {
    const _Float16* Ao = normed_h + (size_t)OTH[i] * BAND;
    gemm(Ao, bridge_kW + i * WSTRIDE, bridge_kb + i * DDIM, kf, 0);
    gemm(Ao, bridge_vW + i * WSTRIDE, bridge_vb + i * DDIM, vf, 0);
    rows_to_f16_kernel<<<MROWS / 8, 256, 0, stream>>>(kf, kn, 1, 1.0f);
    topk_attn_wmma_kernel<<<64, 256, 0, stream>>>(qn, kn, vf, temps, 6, att);
    {
      int n4 = (int)(BAND / 4);
      ew_add4_kernel<<<2048, 256, 0, stream>>>((float4*)attacc, (const float4*)att, n4);
    }
  }
  rows_to_f16_kernel<<<MROWS / 8, 256, 0, stream>>>(attacc, atth, 0, 1.0f / 6.0f);
  gemm(atth, bridge_oW, bridge_ob, outp + (size_t)3 * BAND, 1);
}
